// LSTMCell_58420145160201
// MI455X (gfx1250) — matrix operand order repaired and checked by reading, not compiled, not run on hardware
//
#include <hip/hip_runtime.h>
#include <cstdint>

// ---------------------------------------------------------------------------
// Fused LSTM cell for MI455X (gfx1250, wave32, WMMA).
//   ifgo = h @ W_h + b_h + x @ W_x         (B=4096, D=1024, 4 gates)
//   c' = sig(f)*c + sig(i)*tanh(g);  h' = sig(o)*tanh(c')
// GEMM in f16 WMMA (fp32 accumulate), epilogue fused (ifgo never hits HBM).
// A tiles: async global->LDS copy (ASYNCcnt), fp32 in LDS, cvt at frag read.
// B tiles: VGPR-staged with k<->n transpose + f16 convert into LDS.
// B fragments are software-pipelined (load i+1 issued before WMMA i) so the
// DS latency hides behind matrix-pipe execution instead of issue stalls.
// ---------------------------------------------------------------------------

typedef _Float16 v16h __attribute__((ext_vector_type(16)));
typedef _Float16 h4   __attribute__((ext_vector_type(4)));
typedef _Float16 h2   __attribute__((ext_vector_type(2)));
typedef float    v8f  __attribute__((ext_vector_type(8)));

struct __align__(16) F4 { float v[4]; };

#define DK        1024          // D_IN == D_H
#define NROWS     4096          // batch
#define NGATE     4
#define BM        128           // block rows
#define BN        32            // block cols per gate slab
#define KSTEP     32            // WMMA K
#define NSTEPS    64            // 2 matrices * (1024/32)
#define A_STR     36            // fp32 row stride: 144B, 16B aligned, conflict-free
#define B_STR     36            // f16 row stride: 72B, 8B aligned, conflict-free

__device__ __forceinline__ float sigm_fast(float xv) {
    return 1.0f / (1.0f + __expf(-xv));
}
__device__ __forceinline__ float tanh_fast(float xv) {
    float e = __expf(2.0f * xv);            // +inf / 0 saturate correctly
    return 1.0f - 2.0f / (e + 1.0f);
}

__global__ __launch_bounds__(256)
void lstm_wmma_f16(const float* __restrict__ x,
                   const float* __restrict__ h,
                   const float* __restrict__ c,
                   const float* __restrict__ W_h,
                   const float* __restrict__ b_h,
                   const float* __restrict__ W_x,
                   float* __restrict__ h_next,
                   float* __restrict__ c_next)
{
    __shared__ __align__(16) float    ldsA[2][BM][A_STR];            // fp32 [m][k]
    __shared__ __align__(16) _Float16 ldsB[2][NGATE * BN][B_STR];    // f16  [n][k]

    const int tid  = threadIdx.x;
    const int lane = tid & 31;
    const int wave = tid >> 5;
    const int m0   = blockIdx.y * BM;   // row base
    const int n0   = blockIdx.x * BN;   // col base within each 1024-wide gate slab

    v8f zero = {};
    v8f acc[NGATE * 2];                 // [g*2 + ct]
    #pragma unroll
    for (int i = 0; i < NGATE * 2; ++i)
        acc[i] = zero;

    F4 bPre0[2], bPre1[2];

    // ---- A tile: async global -> LDS copy (fp32, no transpose needed) ----
    auto issue_a = [&](int s, int buf) {
        const float* Asrc = (s < NSTEPS / 2) ? h : x;
        const int    k0   = (s & (NSTEPS / 2 - 1)) * KSTEP;
        #pragma unroll
        for (int t = 0; t < 4; ++t) {                 // 128x32 f32 = 1024 b128 txfers
            int id = tid + t * 256;
            int m  = id >> 3;
            int k4 = (id & 7) << 2;
            const float* gp = Asrc + (size_t)(m0 + m) * DK + k0 + k4;
            uint32_t ldsoff = (uint32_t)(uintptr_t)&ldsA[buf][m][k4];
            asm volatile("global_load_async_to_lds_b128 %0, %1, off"
                         :: "v"(ldsoff), "v"((uint64_t)(uintptr_t)gp)
                         : "memory");
        }
    };

    // ---- B tile: global fp32 loads (k-pairs enable transposed packed stores) ----
    auto stage_load_b = [&](int s) {
        const float* Wsrc = (s < NSTEPS / 2) ? W_h : W_x;
        const int    k0   = (s & (NSTEPS / 2 - 1)) * KSTEP;
        #pragma unroll
        for (int t = 0; t < 2; ++t) {                 // 32x128 fp32
            int id   = tid + t * 256;
            int kp   = id >> 5;                       // k-pair 0..15
            int n4   = (id & 31) << 2;                // local n 0..124
            int gate = n4 >> 5;
            int np   = n4 & 31;
            const float* p = Wsrc + (size_t)(k0 + 2 * kp) * (NGATE * DK)
                                  + (size_t)gate * DK + n0 + np;
            bPre0[t] = *(const F4*)p;                 // row k
            bPre1[t] = *(const F4*)(p + NGATE * DK);  // row k+1
        }
    };

    auto stage_store_b = [&](int buf) {
        #pragma unroll
        for (int t = 0; t < 2; ++t) {
            int id = tid + t * 256;
            int kp = id >> 5;
            int n4 = (id & 31) << 2;
            #pragma unroll
            for (int j = 0; j < 4; ++j) {             // transpose: 4 b32 stores
                h2 hv = { (_Float16)bPre0[t].v[j], (_Float16)bPre1[t].v[j] };
                *(h2*)&ldsB[buf][n4 + j][2 * kp] = hv;
            }
        }
    };

    // ---- fragment loads, matching ISA 16-bit A / B VGPR layouts ----
    // A 16x32: lanes 0-15 hold K 0..7,16..23; lanes 16-31 hold K 8..15,24..31
    // (fp32 in LDS; convert to f16 here — co-executes with WMMA)
    auto load_a = [&](int buf) -> v16h {
        const float* row = &ldsA[buf][wave * 16 + (lane & 15)][0];
        const int kh = (lane >> 4) * 8;
        F4 q0 = *(const F4*)(row + kh + 0);
        F4 q1 = *(const F4*)(row + kh + 4);
        F4 q2 = *(const F4*)(row + 16 + kh + 0);
        F4 q3 = *(const F4*)(row + 16 + kh + 4);
        union { v16h v; h4 cc[4]; } u;
        u.cc[0] = h4{(_Float16)q0.v[0], (_Float16)q0.v[1], (_Float16)q0.v[2], (_Float16)q0.v[3]};
        u.cc[1] = h4{(_Float16)q1.v[0], (_Float16)q1.v[1], (_Float16)q1.v[2], (_Float16)q1.v[3]};
        u.cc[2] = h4{(_Float16)q2.v[0], (_Float16)q2.v[1], (_Float16)q2.v[2], (_Float16)q2.v[3]};
        u.cc[3] = h4{(_Float16)q3.v[0], (_Float16)q3.v[1], (_Float16)q3.v[2], (_Float16)q3.v[3]};
        return u.v;
    };
    // B 32x16: lane n = N, lanes 0-15 hold K 0..7,16..23; lanes 16-31 hold K 8..15,24..31
    auto load_b = [&](int buf, int nloc) -> v16h {
        const _Float16* row = &ldsB[buf][nloc + (lane & 15)][(lane >> 4) * 8];
        union { v16h v; h4 cc[4]; } u;
        u.cc[0] = *(const h4*)(row + 0);
        u.cc[1] = *(const h4*)(row + 4);
        u.cc[2] = *(const h4*)(row + 16);
        u.cc[3] = *(const h4*)(row + 20);
        return u.v;
    };

    // ---- software-pipelined K loop (double-buffered LDS) ----
    issue_a(0, 0);
    stage_load_b(0);
    stage_store_b(0);
    asm volatile("s_wait_asynccnt 0x0" ::: "memory");
    __syncthreads();

    for (int s = 0; s < NSTEPS; ++s) {
        const int buf = s & 1;
        if (s + 1 < NSTEPS) {
            issue_a(s + 1, buf ^ 1);                  // background copy during compute
            stage_load_b(s + 1);
        }

        v16h af = load_a(buf);

        // Software-pipelined B fragments: the load for fragment i+1 is a true
        // producer of the value consumed AFTER WMMA i, forcing two live
        // fragment octets and letting DS latency hide behind the matrix pipe.
        v16h bf_cur = load_b(buf, 0);
        #pragma unroll
        for (int i = 0; i < NGATE * 2; ++i) {
            v16h bf_next = (i + 1 < NGATE * 2) ? load_b(buf, (i + 1) * 16) : bf_cur;
            acc[i] = __builtin_amdgcn_wmma_f32_16x16x32_f16(
                false, af, false, bf_cur, (short)0, acc[i], false, false);
            bf_cur = bf_next;
        }

        if (s + 1 < NSTEPS) stage_store_b(buf ^ 1);
        asm volatile("s_wait_asynccnt 0x0" ::: "memory");
        __syncthreads();
    }

    // ---- fused LSTM epilogue (bias + activations), ifgo never leaves VGPRs ----
    // D-matrix layout: VGPR r, lane l -> M = r + (l>=16)*8, N = l&15
    const int kh   = lane >> 4;
    const int nn   = lane & 15;
    const int rowb = m0 + wave * 16 + kh * 8;

    #pragma unroll
    for (int ct = 0; ct < 2; ++ct) {
        const int col = n0 + ct * 16 + nn;            // 0..1023
        const float bi = b_h[0 * DK + col];
        const float bf = b_h[1 * DK + col];
        const float bg = b_h[2 * DK + col];
        const float bo = b_h[3 * DK + col];
        #pragma unroll
        for (int r = 0; r < 8; ++r) {
            const int    row = rowb + r;
            const size_t idx = (size_t)row * DK + col;
            const float iv = acc[0 * 2 + ct][r] + bi;
            const float fv = acc[1 * 2 + ct][r] + bf;
            const float gv = acc[2 * 2 + ct][r] + bg;
            const float ov = acc[3 * 2 + ct][r] + bo;
            const float cold = c[idx];
            const float cn = sigm_fast(fv) * cold + sigm_fast(iv) * tanh_fast(gv);
            const float hn = sigm_fast(ov) * tanh_fast(cn);
            h_next[idx] = hn;
            c_next[idx] = cn;
        }
    }
}

extern "C" void kernel_launch(void* const* d_in, const int* in_sizes, int n_in,
                              void* d_out, int out_size, void* d_ws, size_t ws_size,
                              hipStream_t stream) {
    const float* x   = (const float*)d_in[0];
    const float* h   = (const float*)d_in[1];
    const float* c   = (const float*)d_in[2];
    const float* W_h = (const float*)d_in[3];
    const float* b_h = (const float*)d_in[4];
    const float* W_x = (const float*)d_in[5];

    float* h_next = (float*)d_out;
    float* c_next = h_next + (size_t)NROWS * DK;

    dim3 grid(DK / BN, NROWS / BM);   // (32, 32)
    lstm_wmma_f16<<<grid, 256, 0, stream>>>(x, h, c, W_h, b_h, W_x, h_next, c_next);
}
